// SupConLossMemory_36172214567215
// MI455X (gfx1250) — matrix operand order repaired and checked by reading, not compiled, not run on hardware
//
#include <hip/hip_runtime.h>
#include <hip/hip_bf16.h>

typedef _Float16 v4h  __attribute__((ext_vector_type(4)));
typedef _Float16 v8h  __attribute__((ext_vector_type(8)));
typedef _Float16 v16h __attribute__((ext_vector_type(16)));
typedef float    v4f  __attribute__((ext_vector_type(4)));
typedef float    v8f  __attribute__((ext_vector_type(8)));

#define S_TOT    16384
#define A_TOT    4096
#define D_DIM    128
#define DIAG_OFF (S_TOT - A_TOT)          /* 12288 */
#define INV_T    14.285714285714286f      /* 1/0.07 */
#define C_SHIFT  14.285714285714286f      /* provable max logit (unit vectors) */
#define LOG2E    1.4426950408889634f
#define LN2      0.6931471805599453f
#define KEXP     (INV_T * LOG2E)          /* exp2(acc*KEXP - KEXP) = exp(x - C) */

// ---------------------------------------------------------------------------
// Kernel 1: fp32 -> fp16 conversion of anchor (features) and contrast
// (memory_bank) into workspace.
// ---------------------------------------------------------------------------
__global__ __launch_bounds__(256)
void convert_f16_kernel(const float* __restrict__ mb, const float* __restrict__ ft,
                        _Float16* __restrict__ con_h, _Float16* __restrict__ anc_h) {
    const size_t ac = (size_t)A_TOT * D_DIM / 4;
    const size_t cc = (size_t)S_TOT * D_DIM / 4;
    size_t i = (size_t)blockIdx.x * blockDim.x + threadIdx.x;
    if (i < ac) {
        v4f x = ((const v4f*)ft)[i];
        v4h y = { (_Float16)x.x, (_Float16)x.y, (_Float16)x.z, (_Float16)x.w };
        ((v4h*)anc_h)[i] = y;
    } else if (i < ac + cc) {
        size_t j = i - ac;
        v4f x = ((const v4f*)mb)[j];
        v4h y = { (_Float16)x.x, (_Float16)x.y, (_Float16)x.z, (_Float16)x.w };
        ((v4h*)con_h)[j] = y;
    }
}

// ---------------------------------------------------------------------------
// Kernel 2: class histogram of the 16384 labels (13 classes) -> denom_i is
// count[lab_i]-1, so positive-count accumulation vanishes from the hot loop.
// ---------------------------------------------------------------------------
__global__ __launch_bounds__(256)
void label_hist_kernel(const int* __restrict__ lab, int* __restrict__ counts) {
    __shared__ int h[16];
    if (threadIdx.x < 16) h[threadIdx.x] = 0;
    __syncthreads();
    for (int i = threadIdx.x; i < S_TOT; i += 256) atomicAdd(&h[lab[i]], 1);
    __syncthreads();
    if (threadIdx.x < 16) counts[threadIdx.x] = h[threadIdx.x];
}

// ---------------------------------------------------------------------------
// Kernel 3: fused GEMM + streaming SupCon statistics.
// Block = 16 anchor rows, 8 waves. A (16x128 f16) pinned in VGPRs. Each wave
// streams 128 column tiles with DOUBLE-BUFFERED B fragments: loads for tile
// t+1 are in flight while 4 WMMAs + epilogue consume tile t. Diagonal terms
// are included unconditionally and subtracted exactly in the reduction phase.
// ---------------------------------------------------------------------------
__global__ __launch_bounds__(256)
void supcon_kernel(const _Float16* __restrict__ anc, const _Float16* __restrict__ con,
                   const int* __restrict__ lab, const int* __restrict__ counts,
                   float* __restrict__ partial) {
    __shared__ float red_e[16][128];
    __shared__ float red_p[16][128];
    __shared__ float rowres[16];

    const int tid  = threadIdx.x;
    const int lane = tid & 31;
    const int wave = tid >> 5;
    const int l15  = lane & 15;
    const int h    = lane >> 4;          // K-subchunk half (A/B), row half (C)
    const int rowbase = blockIdx.x * 16;

    // --- A fragments: lane holds anchor row (rowbase + l15).
    const _Float16* ap = anc + (size_t)(rowbase + l15) * D_DIM;
    v16h afrag[4];
#pragma unroll
    for (int ks = 0; ks < 4; ++ks) {
        v8h lo = *(const v8h*)(ap + ks * 32 + h * 16);
        v8h hi = *(const v8h*)(ap + ks * 32 + h * 16 + 8);
        afrag[ks] = __builtin_shufflevector(lo, hi,
                    0, 1, 2, 3, 4, 5, 6, 7, 8, 9, 10, 11, 12, 13, 14, 15);
    }

    const int growbase = rowbase + h * 8;
    int rlab[8];
#pragma unroll
    for (int v = 0; v < 8; ++v) rlab[v] = lab[DIAG_OFF + growbase + v];

    float esum[8], psum[8];
#pragma unroll
    for (int v = 0; v < 8; ++v) { esum[v] = 0.f; psum[v] = 0.f; }

    // --- Prime the pipeline: tile 0 for this wave.
    int c = wave * 16 + l15;
    const _Float16* bp = con + (size_t)c * D_DIM + h * 16;
    v16h b0 = *(const v16h*)(bp +  0);
    v16h b1 = *(const v16h*)(bp + 32);
    v16h b2 = *(const v16h*)(bp + 64);
    v16h b3 = *(const v16h*)(bp + 96);
    int clab = lab[c];

    for (int t = 0; t < S_TOT / (16 * 8); ++t) {
        // Issue next tile's loads (wrap on last iter; data unused but in-bounds).
        const int cn = (c + 128) & (S_TOT - 1);
        const _Float16* np = con + (size_t)cn * D_DIM + h * 16;
        v16h n0 = *(const v16h*)(np +  0);
        v16h n1 = *(const v16h*)(np + 32);
        v16h n2 = *(const v16h*)(np + 64);
        v16h n3 = *(const v16h*)(np + 96);
        const int nlab = lab[cn];

        // Consume current tile (registers already resident -> no load wait).
        v8f acc = {};
        acc = __builtin_amdgcn_wmma_f32_16x16x32_f16(false, afrag[0], false, b0, (short)0, acc, false, false);
        acc = __builtin_amdgcn_wmma_f32_16x16x32_f16(false, afrag[1], false, b1, (short)0, acc, false, false);
        acc = __builtin_amdgcn_wmma_f32_16x16x32_f16(false, afrag[2], false, b2, (short)0, acc, false, false);
        acc = __builtin_amdgcn_wmma_f32_16x16x32_f16(false, afrag[3], false, b3, (short)0, acc, false, false);

#pragma unroll
        for (int v = 0; v < 8; ++v) {
            float e = __builtin_amdgcn_exp2f(acc[v] * KEXP - KEXP);  // exp(x - C)
            esum[v] += e;                                            // diag fixed later
            bool pos = (clab == rlab[v]);
            psum[v] += pos ? acc[v] : 0.f;                           // raw units
        }

        // Rotate double buffer (this is where the load wait lands).
        b0 = n0; b1 = n1; b2 = n2; b3 = n3; clab = nlab; c = cn;
    }

    // --- Combine 128 partials per row via LDS.
    const int slot = wave * 16 + l15;
#pragma unroll
    for (int v = 0; v < 8; ++v) {
        red_e[h * 8 + v][slot] = esum[v];
        red_p[h * 8 + v][slot] = psum[v];
    }
    __syncthreads();

    if (tid < 16) {
        float es = 0.f, ps = 0.f;
        for (int s = 0; s < 128; ++s) { es += red_e[tid][s]; ps += red_p[tid][s]; }

        const int grow = rowbase + tid;
        // Exact diagonal term: raw f16 dot of anchor row with its contrast row.
        const _Float16* arow = anc + (size_t)grow * D_DIM;
        const _Float16* crow = con + (size_t)(DIAG_OFF + grow) * D_DIM;
        float draw = 0.f;
        for (int k = 0; k < D_DIM; ++k) draw += (float)arow[k] * (float)crow[k];

        es -= __builtin_amdgcn_exp2f(draw * KEXP - KEXP);   // remove diag from exp-sum
        ps  = (ps - draw) * INV_T;                          // remove diag, scale by 1/T

        const float denom = (float)(counts[lab[DIAG_OFF + grow]] - 1);
        const float lse   = __builtin_amdgcn_logf(es) * LN2 + C_SHIFT;
        const float possum = ps - denom * lse;
        const float mean   = possum / (denom > 0.f ? denom : 1.f);
        rowres[tid] = -mean;
    }
    __syncthreads();
    if (tid == 0) {
        float s = 0.f;
        for (int r = 0; r < 16; ++r) s += rowres[r];
        partial[blockIdx.x] = s;
    }
}

// ---------------------------------------------------------------------------
// Kernel 4: fixed-order final reduction (deterministic, no float atomics).
// ---------------------------------------------------------------------------
__global__ __launch_bounds__(256)
void finalize_kernel(const float* __restrict__ partial, float* __restrict__ out) {
    __shared__ float sh[256];
    sh[threadIdx.x] = partial[threadIdx.x];
    __syncthreads();
    if (threadIdx.x == 0) {
        float s = 0.f;
        for (int i = 0; i < 256; ++i) s += sh[i];
        out[0] = s;
    }
}

extern "C" void kernel_launch(void* const* d_in, const int* in_sizes, int n_in,
                              void* d_out, int out_size, void* d_ws, size_t ws_size,
                              hipStream_t stream) {
    (void)in_sizes; (void)n_in; (void)out_size; (void)ws_size;
    const float* mb  = (const float*)d_in[0];   // memory_bank (1,S,D) fp32
    const int*   lab = (const int*)d_in[1];     // labels (S,1) int32
    const float* ft  = (const float*)d_in[2];   // features (1,A,D) fp32
    float* out = (float*)d_out;

    _Float16* anc_h = (_Float16*)d_ws;
    _Float16* con_h = anc_h + (size_t)A_TOT * D_DIM;
    float* partial  = (float*)(con_h + (size_t)S_TOT * D_DIM);
    int*   counts   = (int*)(partial + 256);

    const size_t nv4 = ((size_t)(A_TOT + S_TOT) * D_DIM) / 4;   // 655360
    convert_f16_kernel<<<dim3((unsigned)((nv4 + 255) / 256)), dim3(256), 0, stream>>>(
        mb, ft, con_h, anc_h);
    label_hist_kernel<<<dim3(1), dim3(256), 0, stream>>>(lab, counts);
    supcon_kernel<<<dim3(A_TOT / 16), dim3(256), 0, stream>>>(anc_h, con_h, lab, counts, partial);
    finalize_kernel<<<dim3(1), dim3(256), 0, stream>>>(partial, out);
}
